// WindowAttentionFusion_63075889709435
// MI455X (gfx1250) — compile-verified, matrix operand order checked by reading, not run on hardware
//
#include <hip/hip_runtime.h>
#include <hip/hip_bf16.h>

// ---------------------------------------------------------------------------
// 3D window attention (Swin), window 4x4x4 (N=64), C=192, 6 heads, hd=32.
// x: (2,192,46,47,48) f32. Padded grid 48^3 -> 12^3 windows/batch, 3456 total.
// One workgroup (256 thr = 8 wave32) per window; all GEMMs on
// v_wmma_f32_16x16x32_f16. GEMM orientations chosen so every LDS C-tile
// spill is one ds_store_b128 per lane (M-axis of the WMMA C-layout mapped
// onto the contiguous LDS axis of the destination buffer).
// ---------------------------------------------------------------------------

typedef __attribute__((ext_vector_type(16))) _Float16 v16h;
typedef __attribute__((ext_vector_type(8)))  _Float16 v8h;
typedef __attribute__((ext_vector_type(8)))  float    v8f;

#define DIM      192
#define HEADS    6
#define HD       32
#define NTOK     64
#define D_IN     46
#define H_IN     47
#define W_IN     48
#define NWD      12
#define NWIN_B   (NWD*NWD*NWD)
#define QKV_COLS (3*DIM)                 // 576
#define QKROW    (2*DIM)                 // qkvs row stride (Q|K only) = 384
#define SCALE    0.17677669529663689f    // 32^-0.5

// LDS carve (in _Float16 units; float tail at the end)
#define XF_OFF   0                         // 64*192  = 12288 halfs
#define QKVS_OFF (XF_OFF + NTOK*DIM)       // 64*384  = 24576 halfs (Q|K)
#define VT_OFF   (QKVS_OFF + NTOK*QKROW)   // 192*64  = 12288 halfs
#define PST_OFF  (VT_OFF + DIM*NTOK)       // 8*16*64 =  8192 halfs
#define AO_OFF   (PST_OFF + 8*16*NTOK)     // 64*192  = 12288 halfs
#define HALF_TOT (AO_OFF + NTOK*DIM)       // 69632 halfs = 139264 B
#define RPB_OFF_B  (HALF_TOT*2)
#define QKVB_OFF_B (RPB_OFF_B + 343*HEADS*4)
#define PRJB_OFF_B (QKVB_OFF_B + QKV_COLS*4)
#define SMEM_BYTES (PRJB_OFF_B + DIM*4)    // 150568 B (< 320 KB/WGP)

// 16-bit A/B WMMA fragment: lane holds 16 halfs of a 32-wide K chunk.
// HW layout: lanes 0-15 -> K {0..7,16..23}, lanes 16-31 -> K {8..15,24..31}.
__device__ __forceinline__ v16h load_frag(const _Float16* base32, int half16) {
    const v8h lo = *reinterpret_cast<const v8h*>(base32 + half16 * 8);
    const v8h hi = *reinterpret_cast<const v8h*>(base32 + 16 + half16 * 8);
    v16h r;
#pragma unroll
    for (int i = 0; i < 8; ++i) { r[i] = lo[i]; r[i + 8] = hi[i]; }
    return r;
}

__device__ __forceinline__ v8f wmma_f16(v16h a, v16h b, v8f c) {
    return __builtin_amdgcn_wmma_f32_16x16x32_f16(false, a, false, b,
                                                  (short)0, c, false, false);
}

// ------------------- prologue: fp32 weights -> f16 scratch ------------------
__global__ void wconv_kernel(const float* __restrict__ qkv_w,
                             const float* __restrict__ proj_w,
                             _Float16* __restrict__ qkv_h,
                             _Float16* __restrict__ proj_h) {
    int i = blockIdx.x * 256 + threadIdx.x;
    if (i < QKV_COLS * DIM) qkv_h[i] = (_Float16)qkv_w[i];
    if (i < DIM * DIM)      proj_h[i] = (_Float16)proj_w[i];
}

// ------------------------ fused per-window attention ------------------------
__global__ void __launch_bounds__(256, 1)
win_attn_kernel(const float* __restrict__ x,
                const float* __restrict__ qkv_b,
                const float* __restrict__ proj_b,
                const float* __restrict__ rpb,       // (343, 6)
                const _Float16* __restrict__ qkv_wh, // (576, 192) f16
                const _Float16* __restrict__ proj_wh,// (192, 192) f16
                float* __restrict__ out) {
    extern __shared__ char smem[];
    _Float16* xf   = reinterpret_cast<_Float16*>(smem) + XF_OFF;
    _Float16* qkvs = reinterpret_cast<_Float16*>(smem) + QKVS_OFF; // [tok][Q|K ch]
    _Float16* vT   = reinterpret_cast<_Float16*>(smem) + VT_OFF;   // [v ch][tok]
    _Float16* pst  = reinterpret_cast<_Float16*>(smem) + PST_OFF;  // per-wave [i][j]
    _Float16* ao   = reinterpret_cast<_Float16*>(smem) + AO_OFF;   // [tok][ch]
    float*    rpbl  = reinterpret_cast<float*>(smem + RPB_OFF_B);
    float*    qkvbl = reinterpret_cast<float*>(smem + QKVB_OFF_B);
    float*    prjbl = reinterpret_cast<float*>(smem + PRJB_OFF_B);

    const int tid  = threadIdx.x;
    const int wave = tid >> 5;
    const int lane = tid & 31;
    const int ln   = lane & 15;
    const int half16 = lane >> 4;

    const int win = blockIdx.x;
    const int b   = win / NWIN_B;
    int r0 = win - b * NWIN_B;
    const int dw = r0 / (NWD * NWD);
    const int hw = (r0 / NWD) % NWD;
    const int ww = r0 % NWD;
    const int d0 = dw * 4, h0 = hw * 4, w0 = ww * 4;

    // ---- Phase A: window -> LDS f16 (zero pad); RPB + biases -> LDS -------
    for (int i = tid; i < NTOK * DIM; i += 256) {
        int t = i & 63;            // token fastest: 4 consecutive w coalesce
        int c = i >> 6;
        int d = d0 + (t >> 4), h = h0 + ((t >> 2) & 3), w = w0 + (t & 3);
        float v = 0.0f;
        if (d < D_IN && h < H_IN) {
            long gi = ((((long)(b * DIM + c) * D_IN + d) * H_IN + h) * W_IN + w);
            v = x[gi];
        }
        xf[t * DIM + c] = (_Float16)v;
    }
    for (int i = tid; i < 343 * HEADS; i += 256) rpbl[i] = rpb[i];
    if (tid < QKV_COLS) qkvbl[tid] = qkv_b[tid];          // 576 <= 256? no:
    for (int i = tid + 256; i < QKV_COLS; i += 256) qkvbl[i] = qkv_b[i];
    if (tid < DIM) prjbl[tid] = proj_b[tid];
    __syncthreads();

    // ---- Phase B: QKV GEMM, transposed orientation for contiguous spills --
    // tiles 0..95  : Q|K, C = W.X^T  (rows = out-ch, cols = tok) -> qkvs[tok][ch]
    // tiles 96..143: V,   C = X.W^T  (rows = tok, cols = v-ch)   -> vT[ch][tok]
    for (int tt = wave; tt < 144; tt += 8) {
        if (tt < 96) {
            int mt = tt / 24, nt = tt % 24;          // mt: tok tile, nt: ch tile
            const _Float16* arow = &qkv_wh[(nt * 16 + ln) * DIM];
            const _Float16* brow = &xf[(mt * 16 + ln) * DIM];
            __builtin_prefetch(arow + ln * DIM, 0, 1);
            v8f acc = {};
#pragma unroll
            for (int kk = 0; kk < DIM / 32; ++kk) {
                v16h a = load_frag(arow + kk * 32, half16);
                v16h bw = load_frag(brow + kk * 32, half16);
                acc = wmma_f16(a, bw, acc);
            }
            v8h pk;
#pragma unroll
            for (int r = 0; r < 8; ++r)
                pk[r] = (_Float16)(acc[r] + qkvbl[nt * 16 + 8 * half16 + r]);
            *reinterpret_cast<v8h*>(
                &qkvs[(mt * 16 + ln) * QKROW + nt * 16 + 8 * half16]) = pk;
        } else {
            int v = tt - 96;
            int mt = v / 12, vt = v % 12;            // mt: tok tile, vt: v-ch tile
            const _Float16* arow = &xf[(mt * 16 + ln) * DIM];
            const _Float16* brow = &qkv_wh[(2 * DIM + vt * 16 + ln) * DIM];
            __builtin_prefetch(brow + ln * DIM, 0, 1);
            v8f acc = {};
#pragma unroll
            for (int kk = 0; kk < DIM / 32; ++kk) {
                v16h a = load_frag(arow + kk * 32, half16);
                v16h bw = load_frag(brow + kk * 32, half16);
                acc = wmma_f16(a, bw, acc);
            }
            float bias = qkvbl[2 * DIM + vt * 16 + ln];
            v8h pk;
#pragma unroll
            for (int r = 0; r < 8; ++r) pk[r] = (_Float16)(acc[r] + bias);
            *reinterpret_cast<v8h*>(
                &vT[(vt * 16 + ln) * NTOK + mt * 16 + 8 * half16]) = pk;
        }
    }
    __syncthreads();

    // ---- Phase C: 24 (head,row-block) tasks / 8 waves ---------------------
    // S^T = K.Q^T (rows j = keys, cols i = queries): softmax over j is mostly
    // in-lane (32/64 values) + one shfl_xor(16). P staged row-contiguous,
    // O^T = V^T.P gives channel-major C rows -> b128 stores into ao[tok][ch].
    _Float16* pw = pst + wave * 16 * NTOK;
    for (int task = wave; task < HEADS * 4; task += 8) {
        int h  = task % HEADS;
        int mb = task / HEADS;
        const int i_tok = mb * 16 + ln;              // query token (both halves)
        v16h bq = load_frag(&qkvs[i_tok * QKROW + h * HD], half16);
        v8f s[4];
#pragma unroll
        for (int jt = 0; jt < 4; ++jt) {
            v16h ak = load_frag(&qkvs[(jt * 16 + ln) * QKROW + DIM + h * HD], half16);
            v8f z = {};
            s[jt] = wmma_f16(ak, bq, z);             // rows=j, cols=i
        }
        // scale + relative position bias
        int di_ = i_tok >> 4, hi_ = (i_tok >> 2) & 3, wi_ = i_tok & 3;
#pragma unroll
        for (int jt = 0; jt < 4; ++jt)
#pragma unroll
            for (int r = 0; r < 8; ++r) {
                int j = jt * 16 + r + 8 * half16;
                int dj = j >> 4, hj = (j >> 2) & 3, wj = j & 3;
                int rel = (di_ - dj + 3) * 49 + (hi_ - hj + 3) * 7 + (wi_ - wj + 3);
                s[jt][r] = s[jt][r] * SCALE + rpbl[rel * HEADS + h];
            }
        // softmax over j (column i fixed per ln; halves combined via xor 16)
        float m = s[0][0];
#pragma unroll
        for (int jt = 0; jt < 4; ++jt)
#pragma unroll
            for (int r = 0; r < 8; ++r) m = fmaxf(m, s[jt][r]);
        m = fmaxf(m, __shfl_xor(m, 16, 32));
        float sum = 0.0f;
#pragma unroll
        for (int jt = 0; jt < 4; ++jt)
#pragma unroll
            for (int r = 0; r < 8; ++r) {
                s[jt][r] = __expf(s[jt][r] - m);
                sum += s[jt][r];
            }
        sum += __shfl_xor(sum, 16, 32);
        float inv = 1.0f / sum;
        // stage P row-contiguous: pw[i_local][j], one b128 per jt
#pragma unroll
        for (int jt = 0; jt < 4; ++jt) {
            v8h pk;
#pragma unroll
            for (int r = 0; r < 8; ++r) pk[r] = (_Float16)(s[jt][r] * inv);
            *reinterpret_cast<v8h*>(&pw[ln * NTOK + jt * 16 + 8 * half16]) = pk;
        }
        asm volatile("s_wait_dscnt 0" ::: "memory");  // same-wave LDS RAW
        // O^T = V^T . P  (A: vT channel rows, B: P rows; both K-contiguous)
        v16h bp0 = load_frag(&pw[ln * NTOK], half16);
        v16h bp1 = load_frag(&pw[ln * NTOK + 32], half16);
#pragma unroll
        for (int sub = 0; sub < 2; ++sub) {
            v8f o = {};
            v16h av0 = load_frag(&vT[(h * HD + sub * 16 + ln) * NTOK], half16);
            v16h av1 = load_frag(&vT[(h * HD + sub * 16 + ln) * NTOK + 32], half16);
            o = wmma_f16(av0, bp0, o);
            o = wmma_f16(av1, bp1, o);
            v8h pk;
#pragma unroll
            for (int r = 0; r < 8; ++r) pk[r] = (_Float16)o[r];
            *reinterpret_cast<v8h*>(
                &ao[i_tok * DIM + h * HD + sub * 16 + 8 * half16]) = pk;
        }
    }
    __syncthreads();

    // ---- Phase D: proj GEMM (64x192 @ 192x192) + bias + scatter store -----
    for (int tt = wave; tt < 4 * 12; tt += 8) {
        int mt = tt / 12, nt = tt % 12;
        const _Float16* arow = &ao[(mt * 16 + ln) * DIM];
        const _Float16* brow = &proj_wh[(nt * 16 + ln) * DIM];
        __builtin_prefetch(brow + ln * DIM, 0, 1);
        v8f acc = {};
#pragma unroll
        for (int kk = 0; kk < DIM / 32; ++kk) {
            v16h a = load_frag(arow + kk * 32, half16);
            v16h bw = load_frag(brow + kk * 32, half16);
            acc = wmma_f16(a, bw, acc);
        }
        int oc = nt * 16 + ln;
        float pb = prjbl[oc];
#pragma unroll
        for (int r = 0; r < 8; ++r) {
            int t = mt * 16 + r + 8 * half16;
            int d = d0 + (t >> 4), h = h0 + ((t >> 2) & 3), w = w0 + (t & 3);
            if (d < D_IN && h < H_IN) {
                long gi = ((((long)(b * DIM + oc) * D_IN + d) * H_IN + h) * W_IN + w);
                out[gi] = acc[r] + pb;
            }
        }
    }
}

// ---------------------------------------------------------------------------
extern "C" void kernel_launch(void* const* d_in, const int* in_sizes, int n_in,
                              void* d_out, int out_size, void* d_ws, size_t ws_size,
                              hipStream_t stream) {
    (void)in_sizes; (void)n_in; (void)out_size; (void)ws_size;
    const float* x      = (const float*)d_in[0];
    const float* qkv_w  = (const float*)d_in[1];
    const float* qkv_b  = (const float*)d_in[2];
    const float* proj_w = (const float*)d_in[3];
    const float* proj_b = (const float*)d_in[4];
    const float* rpb    = (const float*)d_in[5];
    float* out = (float*)d_out;

    _Float16* qkv_h  = (_Float16*)d_ws;
    _Float16* proj_h = qkv_h + QKV_COLS * DIM;

    static bool attr_set = false;  // idempotent metadata, not run-state
    if (!attr_set) {
        hipFuncSetAttribute(reinterpret_cast<const void*>(win_attn_kernel),
                            hipFuncAttributeMaxDynamicSharedMemorySize, SMEM_BYTES);
        attr_set = true;
    }

    wconv_kernel<<<(QKV_COLS * DIM + 255) / 256, 256, 0, stream>>>(
        qkv_w, proj_w, qkv_h, proj_h);

    const int nwin = 2 * NWIN_B;  // 3456
    win_attn_kernel<<<nwin, 256, SMEM_BYTES, stream>>>(
        x, qkv_b, proj_b, rpb, qkv_h, proj_h, out);
}
